// MambaBlock_34437047779589
// MI455X (gfx1250) — compile-verified, matrix operand order checked by reading, not run on hardware
//
#include <hip/hip_runtime.h>
#include <hip/hip_bf16.h>
#include <math.h>

// ---------------------------------------------------------------------------
// Mamba block forward for MI455X (gfx1250, wave32, WMMA).
// Pipeline: f32->bf16 converts, WMMA in_proj, conv+SiLU, WMMA x_proj,
// dt_proj+softplus, selective scan (fused gate), WMMA out_proj.
// GEMM: 32x64 wave tile (8 wmma / K-step), N-guard compiled out for the
// large GEMMs so the hot loop has no exec-mask branches.
// ---------------------------------------------------------------------------

typedef __bf16 bf16;
typedef __attribute__((ext_vector_type(16))) __bf16 v16bf;
typedef __attribute__((ext_vector_type(8)))  float  v8f;

#define D_MODEL 1024
#define D_STATE 16
#define D_CONV  4
#define D_INNER 2048
#define DT_RANK 8
#define BATCH   2
#define SEQLEN  1024
#define MROWS   (BATCH * SEQLEN)         // 2048 flattened (b,l) rows
#define XP_COLS (DT_RANK + 2 * D_STATE)  // 40

// ---------------- fragment helpers (layouts per CDNA5 ISA) -------

// A fragment: 16x32 bf16 tile at (m0,k0) of row-major A (ld = K).
// lane<16: row m0+lane, K = {k0+half*8..+7, k0+16+half*8..+7}; half = lane>>4.
__device__ __forceinline__ v16bf load_frag_a(const bf16* __restrict__ A,
                                             int ldA, int m0, int k0, int lane) {
    const int half = lane >> 4, lr = lane & 15;
    const bf16* p = A + (size_t)(m0 + lr) * ldA + (k0 + half * 8);
    v16bf f;
#pragma unroll
    for (int i = 0; i < 8; ++i) { f[i] = p[i]; f[8 + i] = p[16 + i]; }
    return f;
}

// B fragment: 32x16 (KxN) tile at (k0,n0). B[k][n] = W[n][k] with W row-major
// (N,K): column n of B is the contiguous weight row n. Same K striping as A,
// N across lanes. GUARD_N compiles in per-lane OOB zeroing (x_proj, N=40).
template <bool GUARD_N>
__device__ __forceinline__ v16bf load_frag_b(const bf16* __restrict__ W,
                                             int ldW, int n0, int k0, int N,
                                             int lane) {
    const int half = lane >> 4, lr = lane & 15;
    const int n = n0 + lr;
    v16bf f;
    if (!GUARD_N || n < N) {
        const bf16* p = W + (size_t)n * ldW + (k0 + half * 8);
#pragma unroll
        for (int i = 0; i < 8; ++i) { f[i] = p[i]; f[8 + i] = p[16 + i]; }
    } else {
#pragma unroll
        for (int i = 0; i < 16; ++i) f[i] = (bf16)0.0f;
    }
    return f;
}

// C/D store: VGPR r -> M = m0 + r + half*8, N = n0 + (lane&15).
template <bool GUARD_N>
__device__ __forceinline__ void store_tile(float* __restrict__ C, int ldC,
                                           int m0, int n0, int N, int lane,
                                           v8f acc) {
    const int half = lane >> 4, lr = lane & 15;
    const int n = n0 + lr;
    if (!GUARD_N || n < N) {
#pragma unroll
        for (int r = 0; r < 8; ++r)
            C[(size_t)(m0 + r + half * 8) * ldC + n] = acc[r];
    }
}

// ---------------- WMMA GEMM: C[M,N] (f32) = A[M,K] (bf16) * W[N,K]^T (bf16) --
// Block = 4 waves (128 thr), 2x2 wave grid -> 64x128 block tile.
// Wave tile 32x64: 2 A-frags x 4 B-frags = 8 wmma per K-step.
// Grid: (M/64, ceil(N/128)); M, K multiples of 64/32 by construction.
template <bool GUARD_N>
__global__ void __launch_bounds__(128)
gemm_bf16_wmma(const bf16* __restrict__ A, const bf16* __restrict__ W,
               float* __restrict__ C, int M, int N, int K) {
    const int lane = threadIdx.x & 31;
    const int wave = threadIdx.x >> 5;
    const int m0 = blockIdx.x * 64 + (wave >> 1) * 32;
    const int n0 = blockIdx.y * 128 + (wave & 1) * 64;

    v8f zero = {};
    v8f acc[2][4];
#pragma unroll
    for (int i = 0; i < 2; ++i)
#pragma unroll
        for (int j = 0; j < 4; ++j) acc[i][j] = zero;

    for (int k0 = 0; k0 < K; k0 += 32) {
        v16bf a[2], b[4];
#pragma unroll
        for (int i = 0; i < 2; ++i)
            a[i] = load_frag_a(A, K, m0 + 16 * i, k0, lane);
#pragma unroll
        for (int j = 0; j < 4; ++j)
            b[j] = load_frag_b<GUARD_N>(W, K, n0 + 16 * j, k0, N, lane);
#pragma unroll
        for (int i = 0; i < 2; ++i)
#pragma unroll
            for (int j = 0; j < 4; ++j)
                acc[i][j] = __builtin_amdgcn_wmma_f32_16x16x32_bf16(
                    false, a[i], false, b[j], (short)0, acc[i][j], false, false);
    }
#pragma unroll
    for (int i = 0; i < 2; ++i)
#pragma unroll
        for (int j = 0; j < 4; ++j)
            store_tile<GUARD_N>(C, N, m0 + 16 * i, n0 + 16 * j, N, lane,
                                acc[i][j]);
}

// ---------------- f32 -> bf16 convert ---------------------------------------
__global__ void __launch_bounds__(256)
cvt_f32_bf16(const float* __restrict__ in, bf16* __restrict__ out, int n) {
    int i = blockIdx.x * 256 + threadIdx.x;
    if (i < n) out[i] = (bf16)in[i];
}

// ---------------- causal depthwise conv1d + SiLU ----------------------------
// xz rows are 4096 floats: [0,2048)=xb branch, [2048,4096)=gate z.
// xc[b,l,d] = silu( sum_j w[d,j]*xb[b,l-3+j,d] + bias[d] )
__global__ void __launch_bounds__(256)
conv_silu(const float* __restrict__ xz, const float* __restrict__ cw,
          const float* __restrict__ cb, float* __restrict__ xc,
          bf16* __restrict__ xc_bf) {
    int idx = blockIdx.x * 256 + threadIdx.x;          // over MROWS*D_INNER
    int d = idx & (D_INNER - 1);
    int l = (idx >> 11) & (SEQLEN - 1);
    int b = idx >> 21;
    float acc = cb[d];
#pragma unroll
    for (int j = 0; j < D_CONV; ++j) {
        int ll = l - (D_CONV - 1) + j;
        if (ll >= 0)
            acc += cw[d * D_CONV + j] *
                   xz[(size_t)(b * SEQLEN + ll) * (2 * D_INNER) + d];
    }
    float s = acc / (1.0f + __expf(-acc));
    size_t o = (size_t)(b * SEQLEN + l) * D_INNER + d;
    xc[o] = s;
    xc_bf[o] = (bf16)s;
}

// ---------------- dt_proj (K=8) + softplus ----------------------------------
__global__ void __launch_bounds__(256)
delta_softplus(const float* __restrict__ xp, const float* __restrict__ dtw,
               const float* __restrict__ dtb, float* __restrict__ delta) {
    int idx = blockIdx.x * 256 + threadIdx.x;          // over MROWS*D_INNER
    int d = idx & (D_INNER - 1);
    int m = idx >> 11;
    float acc = dtb[d];
    const float* r = xp + (size_t)m * XP_COLS;         // dt_in = xp[0..7]
#pragma unroll
    for (int j = 0; j < DT_RANK; ++j) acc += r[j] * dtw[d * DT_RANK + j];
    float sp = (acc > 20.0f) ? acc : log1pf(__expf(acc));
    delta[(size_t)m * D_INNER + d] = sp;
}

// ---------------- selective scan + output gate ------------------------------
// One lane per (b,d) channel; 16-state h kept in registers; loop over L.
// Fuses y = (scan + D*x) * silu(z), emits bf16 for the out_proj WMMA.
__global__ void __launch_bounds__(256)
selective_scan(const float* __restrict__ xc, const float* __restrict__ delta,
               const float* __restrict__ xp, const float* __restrict__ xz,
               const float* __restrict__ A_log, const float* __restrict__ Dp,
               bf16* __restrict__ ygate_bf) {
    int tid = blockIdx.x * 256 + threadIdx.x;          // 0..4095
    int b = tid >> 11;
    int d = tid & (D_INNER - 1);

    float Av[D_STATE];
#pragma unroll
    for (int n = 0; n < D_STATE; ++n) Av[n] = -__expf(A_log[n]);
    float h[D_STATE];
#pragma unroll
    for (int n = 0; n < D_STATE; ++n) h[n] = 0.0f;
    const float Dd = Dp[d];

    for (int l = 0; l < SEQLEN; ++l) {
        int m = b * SEQLEN + l;
        float xt = xc[(size_t)m * D_INNER + d];
        float dt = delta[(size_t)m * D_INNER + d];
        float dx = dt * xt;
        float y = Dd * xt;
        const float* r = xp + (size_t)m * XP_COLS;     // B = r[8..23], C = r[24..39]
#pragma unroll
        for (int n = 0; n < D_STATE; ++n) {
            float Bt = r[DT_RANK + n];
            float Ct = r[DT_RANK + D_STATE + n];
            h[n] = __expf(dt * Av[n]) * h[n] + dx * Bt;
            y += h[n] * Ct;
        }
        float zv = xz[(size_t)m * (2 * D_INNER) + D_INNER + d];
        float g = zv / (1.0f + __expf(-zv));
        ygate_bf[(size_t)m * D_INNER + d] = (bf16)(y * g);
    }
}

// ---------------------------------------------------------------------------
extern "C" void kernel_launch(void* const* d_in, const int* in_sizes, int n_in,
                              void* d_out, int out_size, void* d_ws, size_t ws_size,
                              hipStream_t stream) {
    const float* x       = (const float*)d_in[0];  // (2,1024,1024)
    const float* w_in    = (const float*)d_in[1];  // (4096,1024)
    const float* conv_w  = (const float*)d_in[2];  // (2048,1,4)
    const float* conv_b  = (const float*)d_in[3];  // (2048)
    const float* w_xp    = (const float*)d_in[4];  // (40,2048)
    const float* dt_w    = (const float*)d_in[5];  // (2048,8)
    const float* dt_b    = (const float*)d_in[6];  // (2048)
    const float* A_log   = (const float*)d_in[7];  // (16)
    const float* D_param = (const float*)d_in[8];  // (2048)
    const float* w_out   = (const float*)d_in[9];  // (1024,2048)
    float* out = (float*)d_out;                    // (2,1024,1024) f32

    // ---- workspace carve-up (~101 MB; whole set L2-resident on 192MB L2) ---
    char* ws = (char*)d_ws;
    size_t off = 0;
    auto carve = [&](size_t bytes) {
        void* p = ws + off;
        off = (off + bytes + 255) & ~(size_t)255;
        return p;
    };
    bf16*  bf_x    = (bf16*) carve((size_t)MROWS * D_MODEL * 2);
    bf16*  bf_win  = (bf16*) carve((size_t)2 * D_INNER * D_MODEL * 2);
    bf16*  bf_wxp  = (bf16*) carve((size_t)XP_COLS * D_INNER * 2);
    bf16*  bf_wout = (bf16*) carve((size_t)D_MODEL * D_INNER * 2);
    bf16*  bf_xc   = (bf16*) carve((size_t)MROWS * D_INNER * 2);
    bf16*  bf_y    = (bf16*) carve((size_t)MROWS * D_INNER * 2);
    float* f_xz    = (float*)carve((size_t)MROWS * 2 * D_INNER * 4);
    float* f_xc    = (float*)carve((size_t)MROWS * D_INNER * 4);
    float* f_xp    = (float*)carve((size_t)MROWS * XP_COLS * 4);
    float* f_delta = (float*)carve((size_t)MROWS * D_INNER * 4);
    (void)ws_size; (void)n_in; (void)in_sizes; (void)out_size;

    // ---- 1) f32 -> bf16 conversions ---------------------------------------
    {
        int n;
        n = MROWS * D_MODEL;
        cvt_f32_bf16<<<(n + 255) / 256, 256, 0, stream>>>(x, bf_x, n);
        n = 2 * D_INNER * D_MODEL;
        cvt_f32_bf16<<<(n + 255) / 256, 256, 0, stream>>>(w_in, bf_win, n);
        n = XP_COLS * D_INNER;
        cvt_f32_bf16<<<(n + 255) / 256, 256, 0, stream>>>(w_xp, bf_wxp, n);
        n = D_MODEL * D_INNER;
        cvt_f32_bf16<<<(n + 255) / 256, 256, 0, stream>>>(w_out, bf_wout, n);
    }

    // ---- 2) in_proj: xz = x @ w_in^T  (2048x1024x4096, WMMA, unguarded) ---
    gemm_bf16_wmma<false>
        <<<dim3(MROWS / 64, (2 * D_INNER) / 128), 128, 0, stream>>>(
            bf_x, bf_win, f_xz, MROWS, 2 * D_INNER, D_MODEL);

    // ---- 3) causal depthwise conv + SiLU ----------------------------------
    conv_silu<<<(MROWS * D_INNER) / 256, 256, 0, stream>>>(
        f_xz, conv_w, conv_b, f_xc, bf_xc);

    // ---- 4) x_proj: xp = xc @ w_xp^T  (N=40, guarded WMMA tiles) ----------
    gemm_bf16_wmma<true>
        <<<dim3(MROWS / 64, (XP_COLS + 127) / 128), 128, 0, stream>>>(
            bf_xc, bf_wxp, f_xp, MROWS, XP_COLS, D_INNER);

    // ---- 5) delta = softplus(dt_in @ dt_w^T + dt_b)  (K=8, VALU) ----------
    delta_softplus<<<(MROWS * D_INNER) / 256, 256, 0, stream>>>(
        f_xp, dt_w, dt_b, f_delta);

    // ---- 6) selective scan + SiLU gate -> bf16 ----------------------------
    selective_scan<<<(BATCH * D_INNER) / 256, 256, 0, stream>>>(
        f_xc, f_delta, f_xp, f_xz, A_log, D_param, bf_y);

    // ---- 7) out_proj: out = ygate @ w_out^T  (2048x2048x1024, WMMA) -------
    gemm_bf16_wmma<false>
        <<<dim3(MROWS / 64, D_MODEL / 128), 128, 0, stream>>>(
            bf_y, bf_wout, out, MROWS, D_MODEL, D_INNER);
}
